// MultiHeadLatentAttentionBlock_1872605741560
// MI455X (gfx1250) — matrix operand order repaired and checked by reading, not compiled, not run on hardware
//
#include <hip/hip_runtime.h>
#include <hip/hip_bf16.h>
#include <math.h>

// ---------------------------------------------------------------------------
// MLA block for MI455X (gfx1250, wave32).
//  - all matmuls via v_wmma_f32_16x16x32_f16 (f32 accumulate)
//  - all LDS tile staging via the Tensor Data Mover (tensor_load_to_lds),
//    double-buffered, waited with s_wait_tensorcnt
//  - f16 operand storage end-to-end (one-time f32->f16 convert/transpose pass)
// B=2, T=2048, D=2048, H=16, DK=128, DC=512, DCR=1024, DR=64
// ---------------------------------------------------------------------------

typedef __attribute__((ext_vector_type(16))) _Float16 v16h;
typedef __attribute__((ext_vector_type(8)))  _Float16 v8h;
typedef __attribute__((ext_vector_type(8)))  float    v8f;
typedef __attribute__((ext_vector_type(4)))  unsigned int v4u;
typedef __attribute__((ext_vector_type(8)))  int      v8i;
typedef __attribute__((ext_vector_type(4)))  int      v4i;

#if __has_builtin(__builtin_amdgcn_tensor_load_to_lds)
#define MLA_TDM 1
#else
#define MLA_TDM 0
#endif

union Frag16 { v16h v; v8h h[2]; };

__device__ __forceinline__ v16h make_frag(const _Float16* lo, const _Float16* hi) {
    Frag16 f;
    f.h[0] = *(const v8h*)lo;
    f.h[1] = *(const v8h*)hi;
    return f.v;
}

__device__ __forceinline__ v8f zero8() {
    v8f z;
    #pragma unroll
    for (int i = 0; i < 8; i++) z[i] = 0.0f;
    return z;
}

// --- TDM 2D tile load: tile (tile_d0 x tile_d1) of f16, row stride stride0
//     elements, from gaddr into LDS at lds byte address. D# per ISA 8.3/8.4.
//     This toolchain exposes the 6-arg builtin:
//     (uint32x4 g0, int32x8 g1, int32x4 g2, int32x4 g3, int32x8 g4, i32 cpol)
__device__ __forceinline__ void tdm_load_2d(unsigned int lds_addr,
                                            const _Float16* gaddr,
                                            unsigned int tensor_d0,
                                            unsigned int tensor_d1,
                                            unsigned int tile_d0,
                                            unsigned int tile_d1,
                                            unsigned long long stride0) {
#if MLA_TDM
    const unsigned long long ga = (unsigned long long)(uintptr_t)gaddr;
    v4u g0;
    g0[0] = 1u;                                            // count=1, user D#
    g0[1] = lds_addr;                                      // lds_addr[31:0]
    g0[2] = (unsigned int)ga;                              // global_addr[31:0]
    g0[3] = ((unsigned int)(ga >> 32) & 0x01FFFFFFu) | 0x80000000u; // addr[56:32] | type=2
    v8i g1;
    g1[0] = 0x00010000;                                    // data_size=1 (2B)
    g1[1] = (int)((tensor_d0 & 0xFFFFu) << 16);            // tensor_dim0[15:0]
    g1[2] = (int)((tensor_d0 >> 16) | ((tensor_d1 & 0xFFFFu) << 16));
    g1[3] = (int)((tensor_d1 >> 16) | (tile_d0 << 16));    // tile_dim0
    g1[4] = (int)(tile_d1 & 0xFFFFu);                      // tile_dim1 (tile_dim2=0)
    g1[5] = (int)(unsigned int)stride0;                    // dim0_stride[31:0]
    g1[6] = (int)((unsigned int)(stride0 >> 32) & 0xFFFFu);
    g1[7] = 0;
    const v4i z4 = {0, 0, 0, 0};
    const v8i z8 = {0, 0, 0, 0, 0, 0, 0, 0};
    __builtin_amdgcn_tensor_load_to_lds(g0, g1, z4, z4, z8, 0);
#else
    (void)lds_addr; (void)gaddr; (void)tensor_d0; (void)tensor_d1;
    (void)tile_d0; (void)tile_d1; (void)stride0;
#endif
}

__device__ __forceinline__ void tdm_wait(int n) {
#if MLA_TDM
#if __has_builtin(__builtin_amdgcn_s_wait_tensorcnt)
    if (n == 0) __builtin_amdgcn_s_wait_tensorcnt(0);
    else        __builtin_amdgcn_s_wait_tensorcnt(2);
#else
    if (n == 0) asm volatile("s_wait_tensorcnt 0x0" ::: "memory");
    else        asm volatile("s_wait_tensorcnt 0x2" ::: "memory");
#endif
#else
    (void)n;
#endif
}

// cooperative fallback copy (all 256 threads), 16B chunks
__device__ __forceinline__ void coop_copy(_Float16* dst, const _Float16* src,
                                          int d0, int d1, long long stride0) {
    const int nch = (d0 * d1) >> 3;
    for (int i = threadIdx.x; i < nch; i += 256) {
        const int lin = i << 3;
        const int r = lin / d0, c = lin % d0;
        *(v8h*)&dst[r * d0 + c] = *(const v8h*)&src[(long long)r * stride0 + c];
    }
}

// ---------------------------------------------------------------------------
// GEMM: C[M,N] = Ah[M,K](f16) @ Wt[N,K](f16, pre-transposed) + bias (f32)
// Output: f16 to Ch (chaining) or f32 to Cf (final). 256 thr, tile 128x128.
// TDM double-buffered K-stream (step 32).
// ---------------------------------------------------------------------------
__global__ __launch_bounds__(256)
void wmma_gemm_tdm(const _Float16* __restrict__ Ah, const _Float16* __restrict__ Wt,
                   const float* __restrict__ bias, float* __restrict__ Cf,
                   _Float16* __restrict__ Ch, int M, int N, int K) {
    __shared__ __align__(16) _Float16 As[2][128 * 32];
    __shared__ __align__(16) _Float16 Bs[2][128 * 32];

    const int tid   = threadIdx.x;
    const int lane  = tid & 31;
    const int w     = tid >> 5;
    const int waveM = (w & 3) * 32;
    const int waveN = (w >> 2) * 64;
    const int m0    = blockIdx.y * 128;
    const int n0    = blockIdx.x * 128;
    const int lrow  = lane & 15;
    const int kg    = lane >> 4;
    const int bRows = (N - n0 < 128) ? (N - n0) : 128;

    v8f acc[2][4];
    #pragma unroll
    for (int i = 0; i < 2; i++)
        #pragma unroll
        for (int j = 0; j < 4; j++) acc[i][j] = zero8();

    const int nt = K >> 5;

#if MLA_TDM
    if (w == 0) {
        tdm_load_2d((unsigned int)(uintptr_t)&As[0][0], Ah + (size_t)m0 * K,
                    K, 128, 32, 128, K);
        tdm_load_2d((unsigned int)(uintptr_t)&Bs[0][0], Wt + (size_t)n0 * K,
                    K, bRows, 32, bRows, K);
    }
#else
    coop_copy(&As[0][0], Ah + (size_t)m0 * K, 32, 128, K);
    coop_copy(&Bs[0][0], Wt + (size_t)n0 * K, 32, bRows, K);
#endif

    for (int t = 0; t < nt; t++) {
        const int buf = t & 1;
#if MLA_TDM
        if (w == 0) {
            if (t + 1 < nt) {
                tdm_load_2d((unsigned int)(uintptr_t)&As[buf ^ 1][0],
                            Ah + (size_t)m0 * K + (size_t)(t + 1) * 32,
                            K, 128, 32, 128, K);
                tdm_load_2d((unsigned int)(uintptr_t)&Bs[buf ^ 1][0],
                            Wt + (size_t)n0 * K + (size_t)(t + 1) * 32,
                            K, bRows, 32, bRows, K);
                tdm_wait(2);            // current tile done, next 2 in flight
            } else {
                tdm_wait(0);
            }
        }
#else
        if (t + 1 < nt) {
            coop_copy(&As[buf ^ 1][0], Ah + (size_t)m0 * K + (size_t)(t + 1) * 32,
                      32, 128, K);
            coop_copy(&Bs[buf ^ 1][0], Wt + (size_t)n0 * K + (size_t)(t + 1) * 32,
                      32, bRows, K);
        }
#endif
        __syncthreads();

        v16h afr[2], bfr[4];
        #pragma unroll
        for (int mi = 0; mi < 2; mi++) {
            const _Float16* a0 = &As[buf][(waveM + mi * 16 + lrow) * 32];
            afr[mi] = make_frag(a0 + 8 * kg, a0 + 16 + 8 * kg);
        }
        #pragma unroll
        for (int ni = 0; ni < 4; ni++) {
            const _Float16* b0 = &Bs[buf][(waveN + ni * 16 + lrow) * 32];
            bfr[ni] = make_frag(b0 + 8 * kg, b0 + 16 + 8 * kg);
        }
        #pragma unroll
        for (int mi = 0; mi < 2; mi++)
            #pragma unroll
            for (int ni = 0; ni < 4; ni++)
                acc[mi][ni] = __builtin_amdgcn_wmma_f32_16x16x32_f16(
                    false, afr[mi], false, bfr[ni], (short)0, acc[mi][ni],
                    false, false);
        __syncthreads();
    }

    #pragma unroll
    for (int mi = 0; mi < 2; mi++)
        #pragma unroll
        for (int ni = 0; ni < 4; ni++)
            #pragma unroll
            for (int r = 0; r < 8; r++) {
                const int gm = m0 + waveM + mi * 16 + r + 8 * kg;
                const int gn = n0 + waveN + ni * 16 + lrow;
                if (gn < N) {
                    const float vv = acc[mi][ni][r] + bias[gn];
                    if (Ch) Ch[(size_t)gm * N + gn] = (_Float16)vv;
                    else    Cf[(size_t)gm * N + gn] = vv;
                }
            }
}

// ---------------------------------------------------------------------------
// elementwise f32 -> f16
__global__ __launch_bounds__(256)
void cvt_f16(const float* __restrict__ src, _Float16* __restrict__ dst, int n) {
    for (int i = blockIdx.x * 256 + threadIdx.x; i < n; i += gridDim.x * 256)
        dst[i] = (_Float16)src[i];
}

// W[K,N] f32 -> Wt[N,K] f16
__global__ __launch_bounds__(256)
void transpose_cvt(const float* __restrict__ W, _Float16* __restrict__ Wt,
                   int K, int N) {
    const int total = K * N;
    for (int i = blockIdx.x * 256 + threadIdx.x; i < total; i += gridDim.x * 256) {
        const int k = i / N, n = i - k * N;
        Wt[(size_t)n * K + k] = (_Float16)W[i];
    }
}

// ---------------------------------------------------------------------------
// Assemble q (pre-scaled by 1/sqrt(128)) / k as [B*H, T, 128] and v
// TRANSPOSED as [B*H, 128, T] (so the attention V tile is a plain TDM copy).
// Decoupled partial RoPE: angle_j = t[j]*freq_j (j<32), out=[sin, cos].
// ---------------------------------------------------------------------------
__global__ __launch_bounds__(256)
void assemble_qkv(const _Float16* __restrict__ qC, const _Float16* __restrict__ kC,
                  const _Float16* __restrict__ vC, const _Float16* __restrict__ qR,
                  const _Float16* __restrict__ kR,
                  _Float16* __restrict__ q, _Float16* __restrict__ k,
                  _Float16* __restrict__ vT) {
    const int bt = blockIdx.x;        // b*T + t
    const int b  = bt >> 11;
    const int t  = bt & 2047;
    const size_t src_bt = (size_t)bt * 2048;
    const float  scale  = 0.08838834764831845f;   // 1/sqrt(128)

    for (int e = threadIdx.x; e < 2048; e += 256) {
        const int h = e >> 7;
        const int d = e & 127;
        const size_t bh = (size_t)(b * 16 + h);
        float qv, kv;
        if (d < 64) {
            qv = (float)qC[src_bt + h * 128 + d];
            kv = (float)kC[src_bt + h * 128 + d];
        } else {
            const int j = (d - 64) & 31;
            const float freq = __powf(10000.0f, -(float)j * (1.0f / 32.0f));
            const float qa = (float)qR[(size_t)bt * 1024 + h * 64 + j] * freq;
            const float ka = (float)kR[(size_t)bt * 64 + j] * freq;
            if (d < 96) { qv = __sinf(qa); kv = __sinf(ka); }
            else        { qv = __cosf(qa); kv = __cosf(ka); }
        }
        q[(bh * 2048 + t) * 128 + d]  = (_Float16)(qv * scale);
        k[(bh * 2048 + t) * 128 + d]  = (_Float16)kv;
        vT[(bh * 128 + d) * 2048 + t] = (_Float16)vC[src_bt + h * 128 + d];
    }
}

// ---------------------------------------------------------------------------
// Flash attention, causal. One WG per (bh, 128 q rows); 8 waves, wave w owns
// 16 q rows. KV streamed in 32-row blocks, TDM double-buffered.
// ---------------------------------------------------------------------------
__global__ __launch_bounds__(256)
void mla_flash_attn(const _Float16* __restrict__ Qg, const _Float16* __restrict__ Kg,
                    const _Float16* __restrict__ VTg, _Float16* __restrict__ Og) {
    __shared__ __align__(16) _Float16 Qs[128 * 128];      // [t][d]
    __shared__ __align__(16) _Float16 Ks[2][32 * 128];    // [kv][d]
    __shared__ __align__(16) _Float16 Vts[2][128 * 32];   // [d][kv]
    __shared__ __align__(16) _Float16 Ps[128 * 32];       // [t][kv]

    const int bh   = blockIdx.y;
    const int q0   = blockIdx.x * 128;
    const int tid  = threadIdx.x;
    const int lane = tid & 31;
    const int w    = tid >> 5;
    const int lrow = lane & 15;
    const int kg   = lane >> 4;

    const size_t base  = (size_t)bh * 2048 * 128;   // q/k base
    const size_t vbase = (size_t)bh * 128 * 2048;   // vT base

    const int nkb = (q0 + 128) / 32;

#if MLA_TDM
    if (w == 0) {
        tdm_load_2d((unsigned int)(uintptr_t)&Qs[0], Qg + base + (size_t)q0 * 128,
                    128, 128, 128, 128, 128);
        tdm_load_2d((unsigned int)(uintptr_t)&Ks[0][0], Kg + base,
                    128, 32, 128, 32, 128);
        tdm_load_2d((unsigned int)(uintptr_t)&Vts[0][0], VTg + vbase,
                    2048, 128, 32, 128, 2048);
    }
#else
    coop_copy(&Qs[0], Qg + base + (size_t)q0 * 128, 128, 128, 128);
    coop_copy(&Ks[0][0], Kg + base, 128, 32, 128);
    coop_copy(&Vts[0][0], VTg + vbase, 32, 128, 2048);
#endif

    v8f o[8];
    #pragma unroll
    for (int i = 0; i < 8; i++) o[i] = zero8();
    float m_i[8], l_i[8];
    #pragma unroll
    for (int r = 0; r < 8; r++) { m_i[r] = -3.0e38f; l_i[r] = 0.0f; }

    for (int kb = 0; kb < nkb; kb++) {
        const int k0  = kb * 32;
        const int buf = kb & 1;
#if MLA_TDM
        if (w == 0) {
            if (kb + 1 < nkb) {
                tdm_load_2d((unsigned int)(uintptr_t)&Ks[buf ^ 1][0],
                            Kg + base + (size_t)(k0 + 32) * 128,
                            128, 32, 128, 32, 128);
                tdm_load_2d((unsigned int)(uintptr_t)&Vts[buf ^ 1][0],
                            VTg + vbase + (size_t)(k0 + 32),
                            2048, 128, 32, 128, 2048);
                tdm_wait(2);
            } else {
                tdm_wait(0);
            }
        }
#else
        if (kb + 1 < nkb) {
            coop_copy(&Ks[buf ^ 1][0], Kg + base + (size_t)(k0 + 32) * 128,
                      128, 32, 128);
            coop_copy(&Vts[buf ^ 1][0], VTg + vbase + (size_t)(k0 + 32),
                      32, 128, 2048);
        }
#endif
        __syncthreads();

        // S = Q K^T : wave-level 16x32, contraction over d=128
        v8f s[2];
        s[0] = zero8(); s[1] = zero8();
        #pragma unroll
        for (int ks = 0; ks < 4; ks++) {
            const _Float16* ab = &Qs[(w * 16 + lrow) * 128 + ks * 32];
            v16h afr = make_frag(ab + 8 * kg, ab + 16 + 8 * kg);
            #pragma unroll
            for (int nt = 0; nt < 2; nt++) {
                const _Float16* bb = &Ks[buf][(nt * 16 + lrow) * 128 + ks * 32];
                v16h bfr = make_frag(bb + 8 * kg, bb + 16 + 8 * kg);
                s[nt] = __builtin_amdgcn_wmma_f32_16x16x32_f16(
                    false, afr, false, bfr, (short)0, s[nt], false, false);
            }
        }

        // causal mask + online softmax (a row's 16 cols live in one 16-lane half)
        const int qrow_base = q0 + w * 16 + 8 * kg;
        #pragma unroll
        for (int r = 0; r < 8; r++) {
            const int qrow = qrow_base + r;
            float s0 = s[0][r], s1 = s[1][r];
            if (k0 + lrow > qrow)      s0 = -3.0e38f;
            if (k0 + 16 + lrow > qrow) s1 = -3.0e38f;
            float mv = fmaxf(s0, s1);
            #pragma unroll
            for (int off = 1; off < 16; off <<= 1)
                mv = fmaxf(mv, __shfl_xor(mv, off, 32));
            const float mnew  = fmaxf(m_i[r], mv);
            const float alpha = __expf(m_i[r] - mnew);
            const float p0 = __expf(s0 - mnew);
            const float p1 = __expf(s1 - mnew);
            float ps = p0 + p1;
            #pragma unroll
            for (int off = 1; off < 16; off <<= 1)
                ps += __shfl_xor(ps, off, 32);
            l_i[r] = l_i[r] * alpha + ps;
            m_i[r] = mnew;
            #pragma unroll
            for (int nt2 = 0; nt2 < 8; nt2++) o[nt2][r] *= alpha;
            Ps[(w * 16 + r + 8 * kg) * 32 + lrow]      = (_Float16)p0;
            Ps[(w * 16 + r + 8 * kg) * 32 + 16 + lrow] = (_Float16)p1;
        }
        __syncthreads();

        // O += P @ V
        {
            const _Float16* ab = &Ps[(w * 16 + lrow) * 32];
            v16h pfr = make_frag(ab + 8 * kg, ab + 16 + 8 * kg);
            #pragma unroll
            for (int nt2 = 0; nt2 < 8; nt2++) {
                const _Float16* bb = &Vts[buf][(nt2 * 16 + lrow) * 32];
                v16h vfr = make_frag(bb + 8 * kg, bb + 16 + 8 * kg);
                o[nt2] = __builtin_amdgcn_wmma_f32_16x16x32_f16(
                    false, pfr, false, vfr, (short)0, o[nt2], false, false);
            }
        }
        __syncthreads();
    }

    // write out (f16) in [B, T, H*128] layout
    const int b = bh >> 4;
    const int h = bh & 15;
    #pragma unroll
    for (int r = 0; r < 8; r++) {
        const int t = q0 + w * 16 + r + 8 * kg;
        const float inv = 1.0f / l_i[r];
        const size_t off = ((size_t)(b * 2048 + t) * 2048) + h * 128;
        #pragma unroll
        for (int nt2 = 0; nt2 < 8; nt2++)
            Og[off + nt2 * 16 + lrow] = (_Float16)(o[nt2][r] * inv);
    }
}

// ---------------------------------------------------------------------------
extern "C" void kernel_launch(void* const* d_in, const int* in_sizes, int n_in,
                              void* d_out, int out_size, void* d_ws, size_t ws_size,
                              hipStream_t stream) {
    (void)in_sizes; (void)n_in; (void)out_size; (void)ws_size;
    const float* x     = (const float*)d_in[0];
    // d_in[1] = mask (causal, hard-coded)
    const float* W_DKV = (const float*)d_in[2];  const float* b_DKV = (const float*)d_in[3];
    const float* W_UK  = (const float*)d_in[4];  const float* b_UK  = (const float*)d_in[5];
    const float* W_UV  = (const float*)d_in[6];  const float* b_UV  = (const float*)d_in[7];
    const float* W_DQ  = (const float*)d_in[8];  const float* b_DQ  = (const float*)d_in[9];
    const float* W_UQ  = (const float*)d_in[10]; const float* b_UQ  = (const float*)d_in[11];
    const float* W_QR  = (const float*)d_in[12]; const float* b_QR  = (const float*)d_in[13];
    const float* W_KR  = (const float*)d_in[14]; const float* b_KR  = (const float*)d_in[15];
    const float* W_O   = (const float*)d_in[16]; const float* b_O   = (const float*)d_in[17];

    _Float16* hws = (_Float16*)d_ws;
    _Float16* xh    = hws;                   // 4096*2048
    _Float16* wDKVt = hws +  8388608ull;     // [512][2048]
    _Float16* wUKt  = hws +  9437184ull;     // [2048][512]
    _Float16* wUVt  = hws + 10485760ull;     // [2048][512]
    _Float16* wDQt  = hws + 11534336ull;     // [1024][2048]
    _Float16* wUQt  = hws + 13631488ull;     // [2048][1024]
    _Float16* wQRt  = hws + 15728640ull;     // [1024][1024]
    _Float16* wKRt  = hws + 16777216ull;     // [64][2048]
    _Float16* wOt   = hws + 16908288ull;     // [2048][2048]
    _Float16* cKVh  = hws + 21102592ull;     // [4096][512]
    _Float16* cQh   = hws + 23199744ull;     // [4096][1024]
    _Float16* kCh   = hws + 27394048ull;     // [4096][2048]
    _Float16* qCh   = hws + 35782656ull;     // [4096][2048]
    _Float16* qRh   = hws + 44171264ull;     // [4096][1024]
    _Float16* kRh   = hws + 48365568ull;     // [4096][64]
    _Float16* vCh   = hws + 48627712ull;     // [4096][2048]
    _Float16* qh    = hws + 57016320ull;     // [32][2048][128]
    _Float16* kh    = hws + 65404928ull;     // [32][2048][128]
    _Float16* vTh   = hws + 73793536ull;     // [32][128][2048]
    _Float16* atth  = hws + 82182144ull;     // [4096][2048]

    const dim3 blk(256);
    auto gemm_h = [&](const _Float16* A, const _Float16* Wt, const float* bias,
                      _Float16* C, int M, int N, int K) {
        dim3 grid((N + 127) / 128, (M + 127) / 128);
        wmma_gemm_tdm<<<grid, blk, 0, stream>>>(A, Wt, bias, (float*)nullptr, C, M, N, K);
    };

    // one-time f16 conversion / weight transpose pass
    cvt_f16<<<dim3(1024), blk, 0, stream>>>(x, xh, 4096 * 2048);
    transpose_cvt<<<dim3(512), blk, 0, stream>>>(W_DKV, wDKVt, 2048, 512);
    transpose_cvt<<<dim3(512), blk, 0, stream>>>(W_UK,  wUKt,  512, 2048);
    transpose_cvt<<<dim3(512), blk, 0, stream>>>(W_UV,  wUVt,  512, 2048);
    transpose_cvt<<<dim3(512), blk, 0, stream>>>(W_DQ,  wDQt,  2048, 1024);
    transpose_cvt<<<dim3(512), blk, 0, stream>>>(W_UQ,  wUQt,  1024, 2048);
    transpose_cvt<<<dim3(512), blk, 0, stream>>>(W_QR,  wQRt,  1024, 1024);
    transpose_cvt<<<dim3(128), blk, 0, stream>>>(W_KR,  wKRt,  2048, 64);
    transpose_cvt<<<dim3(1024), blk, 0, stream>>>(W_O,  wOt,   2048, 2048);

    gemm_h(xh,   wDKVt, b_DKV, cKVh, 4096,  512, 2048);   // c_kv
    gemm_h(cKVh, wUKt,  b_UK,  kCh,  4096, 2048,  512);   // k_C
    gemm_h(cKVh, wUVt,  b_UV,  vCh,  4096, 2048,  512);   // v_C
    gemm_h(xh,   wDQt,  b_DQ,  cQh,  4096, 1024, 2048);   // c_q
    gemm_h(cQh,  wUQt,  b_UQ,  qCh,  4096, 2048, 1024);   // q_C
    gemm_h(cQh,  wQRt,  b_QR,  qRh,  4096, 1024, 1024);   // q_R (pre-rope)
    gemm_h(xh,   wKRt,  b_KR,  kRh,  4096,   64, 2048);   // k_R (pre-rope)

    assemble_qkv<<<dim3(4096), blk, 0, stream>>>(qCh, kCh, vCh, qRh, kRh, qh, kh, vTh);
    mla_flash_attn<<<dim3(16, 32), blk, 0, stream>>>(qh, kh, vTh, atth);

    // final projection, fp32 output
    dim3 grid(16, 32);
    wmma_gemm_tdm<<<grid, blk, 0, stream>>>(atth, wOt, b_O, (float*)d_out,
                                            (_Float16*)nullptr, 4096, 2048, 2048);
}
